// BSRBF_KANLayer_7078106103839
// MI455X (gfx1250) — matrix operand order repaired and checked by reading, not compiled, not run on hardware
//
#include <hip/hip_runtime.h>

typedef unsigned short u16;
typedef unsigned int   u32;
typedef __attribute__((ext_vector_type(16))) __bf16 v16bf;
typedef __attribute__((ext_vector_type(8)))  float  v8f;

#define IN_DIM 512
#define NBASIS 8
#define KTOT   4608   /* 512 (relu base) + 512*8 (spline+rbf) */
#define LDA    72     /* LDS row stride in halves (64 + 8 pad) */

struct Q2 { uint4 a, b; };   // 32 bytes == v16bf

__device__ __forceinline__ u16 f2bf(float f) {
  u32 u = __float_as_uint(f);
  u32 r = u + 0x7FFFu + ((u >> 16) & 1u);   // round-to-nearest-even
  return (u16)(r >> 16);
}

// CDNA5 async memory->LDS copy (ASYNCcnt-tracked); GV addressing mode.
__device__ __forceinline__ void async_copy_b128(u32 lds_byte_addr, const void* gptr) {
  asm volatile("global_load_async_to_lds_b128 %0, %1, off"
               :: "v"(lds_byte_addr), "v"(gptr) : "memory");
}
__device__ __forceinline__ void wait_async0() {
  asm volatile("s_wait_asynccnt 0x0" ::: "memory");
}

// ---------------- Pass 0: fp32 weights -> concatenated bf16 W' (512 x 4608) --
__global__ __launch_bounds__(256) void bsrbf_convert_weights(
    const float* __restrict__ bw, const float* __restrict__ sw,
    u16* __restrict__ wc) {
  int idx = blockIdx.x * 256 + threadIdx.x;
  if (idx >= 512 * KTOT) return;
  int n = idx / KTOT;
  int k = idx - n * KTOT;
  float v = (k < 512) ? bw[n * 512 + k] : sw[(size_t)n * 4096 + (k - 512)];
  wc[idx] = f2bf(v);
}

// ---------------- Pass 1: LayerNorm + relu / b-spline / rbf -> bf16 act rows -
__global__ __launch_bounds__(256) void bsrbf_act(
    const float* __restrict__ x, const float* __restrict__ gamma,
    const float* __restrict__ beta, u16* __restrict__ act) {
  int row = blockIdx.x;
  int t = threadIdx.x;
  int lane = t & 31, wave = t >> 5;
  const float* xr = x + (size_t)row * IN_DIM;
  float x0 = xr[t], x1 = xr[t + 256];

  // wave32 butterfly reduction, then 8-wave exchange through LDS
  float s  = x0 + x1;
  float sq = x0 * x0 + x1 * x1;
  #pragma unroll
  for (int m = 16; m > 0; m >>= 1) {
    s  += __shfl_xor(s, m, 32);
    sq += __shfl_xor(sq, m, 32);
  }
  __shared__ float2 wred[8];
  if (lane == 0) wred[wave] = make_float2(s, sq);
  __syncthreads();
  float ts = 0.f, tq = 0.f;
  #pragma unroll
  for (int w = 0; w < 8; ++w) { ts += wred[w].x; tq += wred[w].y; }

  float mu   = ts * (1.0f / 512.0f);
  float var  = tq * (1.0f / 512.0f) - mu * mu;
  float rstd = rsqrtf(var + 1e-5f);

  u16* arow = act + (size_t)row * KTOT;

  #pragma unroll
  for (int half = 0; half < 2; ++half) {
    int d = t + half * 256;
    float xv = half ? x1 : x0;
    float xn = (xv - mu) * rstd * gamma[d] + beta[d];

    // relu base part (K index d)
    arow[d] = f2bf(fmaxf(xn, 0.0f));

    // B-spline bases: uniform knots knot(i) = -3.3 + 0.6*i, order 3 -> 8 bases
    float b[11];
    #pragma unroll
    for (int i = 0; i < 11; ++i) {
      float k0 = -3.3f + 0.6f * (float)i;
      b[i] = (xn >= k0 && xn < (k0 + 0.6f)) ? 1.0f : 0.0f;
    }
    #pragma unroll
    for (int k = 1; k <= 3; ++k) {
      float inv = 1.0f / (0.6f * (float)k);
      #pragma unroll
      for (int i = 0; i <= 10 - k; ++i) {
        float kl = -3.3f + 0.6f * (float)i;
        float kr = -3.3f + 0.6f * (float)(i + k + 1);
        b[i] = ((xn - kl) * b[i] + (kr - xn) * b[i + 1]) * inv;
      }
    }
    // RBF part: exp(-((xn - rg[j]) * 7/3)^2), rg[j] = -1.5 + j*3/7
    float v[NBASIS];
    #pragma unroll
    for (int j = 0; j < NBASIS; ++j) {
      float c = -1.5f + (float)j * (3.0f / 7.0f);
      float ss = (xn - c) * (7.0f / 3.0f);
      v[j] = b[j] + __expf(-ss * ss);
    }
    uint4 pk;
    pk.x = (u32)f2bf(v[0]) | ((u32)f2bf(v[1]) << 16);
    pk.y = (u32)f2bf(v[2]) | ((u32)f2bf(v[3]) << 16);
    pk.z = (u32)f2bf(v[4]) | ((u32)f2bf(v[5]) << 16);
    pk.w = (u32)f2bf(v[6]) | ((u32)f2bf(v[7]) << 16);
    *(uint4*)&arow[512 + 8 * d] = pk;   // 16B-aligned (row base 9216B, off 1024+16d)
  }
}

// ---------------- Pass 2: WMMA bf16 GEMM  out[M,512] = A[M,4608] * W'^T -------
// block = 256 threads = 8 waves; block tile 128(M) x 128(N); wave tile 32x64.
// Async double-buffered LDS pipeline: one barrier per 64-wide K chunk.
__global__ __launch_bounds__(256) void bsrbf_gemm(
    const u16* __restrict__ A, const u16* __restrict__ W,
    float* __restrict__ out) {
  __shared__ u16 As[2 * 128 * LDA];
  __shared__ u16 Bs[2 * 128 * LDA];

  int t = threadIdx.x;
  int lane = t & 31, wave = t >> 5;
  int wm = wave & 3;          // 4 waves along M (4 * 32 = 128)
  int wn = wave >> 2;         // 2 waves along N (2 * 64 = 128)
  size_t mbase = (size_t)blockIdx.x * 128;
  int nbase = blockIdx.y * 128;

  v8f acc[2][4] = {};

  // staging: 2 threads per row, 64 B (4 x b128) each, for both A and B tiles
  int sr = t >> 1, sh = t & 1;
  const u16* agp = A + (mbase + sr) * KTOT + sh * 32;
  const u16* bgp = W + (size_t)(nbase + sr) * KTOT + sh * 32;
  u32 aLds = (u32)(size_t)&As[sr * LDA + sh * 32];
  u32 bLds = (u32)(size_t)&Bs[sr * LDA + sh * 32];
  const u32 BUFA = 128 * LDA * 2;   // bytes per buffer

  // prologue: fill buffer 0
  #pragma unroll
  for (int i = 0; i < 4; ++i) {
    async_copy_b128(aLds + i * 16, agp + i * 8);
    async_copy_b128(bLds + i * 16, bgp + i * 8);
  }

  for (int kc = 0; kc < KTOT; kc += 64) {
    int buf = (kc >> 6) & 1;
    wait_async0();        // this wave's copies into `buf` have landed in LDS
    __syncthreads();      // everyone's copies landed; prev compute on buf^1 done

    if (kc + 64 < KTOT) { // issue next chunk into the other buffer (overlaps math)
      const u16* an = agp + kc + 64;
      const u16* bn = bgp + kc + 64;
      u32 al = aLds + (buf ? 0 : BUFA);
      u32 bl = bLds + (buf ? 0 : BUFA);
      #pragma unroll
      for (int i = 0; i < 4; ++i) {
        async_copy_b128(al + i * 16, an + i * 8);
        async_copy_b128(bl + i * 16, bn + i * 8);
      }
    }

    const u16* asb = &As[buf * 128 * LDA];
    const u16* bsb = &Bs[buf * 128 * LDA];

    #pragma unroll
    for (int ks = 0; ks < 64; ks += 32) {
      // A fragment: lane<16 holds K {ks..ks+7, ks+16..ks+23}; lane>=16 +8
      int kbA = ks + ((lane < 16) ? 0 : 8);
      // B fragment: lane<16 holds K {ks..ks+7, ks+16..ks+23}; lane>=16 +8
      int kbB = ks + ((lane < 16) ? 0 : 8);
      int l16 = lane & 15;

      v16bf afr[2], bfr[4];
      #pragma unroll
      for (int i = 0; i < 2; ++i) {
        const u16* ap = &asb[(wm * 32 + i * 16 + l16) * LDA + kbA];
        Q2 qa; qa.a = *(const uint4*)(ap); qa.b = *(const uint4*)(ap + 16);
        afr[i] = __builtin_bit_cast(v16bf, qa);
      }
      #pragma unroll
      for (int j = 0; j < 4; ++j) {
        const u16* bp = &bsb[(wn * 64 + j * 16 + l16) * LDA + kbB];
        Q2 qb; qb.a = *(const uint4*)(bp); qb.b = *(const uint4*)(bp + 16);
        bfr[j] = __builtin_bit_cast(v16bf, qb);
      }
      #pragma unroll
      for (int i = 0; i < 2; ++i)
        #pragma unroll
        for (int j = 0; j < 4; ++j)
          acc[i][j] = __builtin_amdgcn_wmma_f32_16x16x32_bf16(
              false, afr[i], false, bfr[j], (short)0, acc[i][j], false, false);
    }
  }

  // Epilogue: 16x16 f32 C/D layout — VGPR r: lanes0-15 M=r, lanes16-31 M=r+8
  int nIn  = lane & 15;
  int mAdd = (lane < 16) ? 0 : 8;
  #pragma unroll
  for (int i = 0; i < 2; ++i)
    #pragma unroll
    for (int j = 0; j < 4; ++j) {
      size_t mrow = mbase + wm * 32 + i * 16 + mAdd;
      int col = nbase + wn * 64 + j * 16 + nIn;
      #pragma unroll
      for (int r = 0; r < 8; ++r)
        out[(mrow + r) * 512 + col] = acc[i][j][r];
    }
}

extern "C" void kernel_launch(void* const* d_in, const int* in_sizes, int n_in,
                              void* d_out, int out_size, void* d_ws, size_t ws_size,
                              hipStream_t stream) {
  const float* x  = (const float*)d_in[0];
  const float* g  = (const float*)d_in[1];
  const float* b  = (const float*)d_in[2];
  const float* bw = (const float*)d_in[3];
  const float* sw = (const float*)d_in[4];
  // d_in[5] (grid) and d_in[6] (rbf_grid) are uniform; reproduced analytically.
  float* out = (float*)d_out;

  long long M = (long long)in_sizes[0] / IN_DIM;   // 8*2048 = 16384

  u16* wc = (u16*)d_ws;
  size_t wbytes = (size_t)512 * KTOT * sizeof(u16);          // 4.5 MiB
  u16* act = (u16*)((char*)d_ws + wbytes);
  size_t avail = (ws_size > wbytes) ? (ws_size - wbytes) : 0;
  long long rows = (long long)(avail / ((size_t)KTOT * sizeof(u16)));
  rows = (rows / 128) * 128;           // GEMM M tile = 128
  if (rows <= 0) rows = 128;           // minimal chunk (~1.2 MiB)
  if (rows > 8192) rows = 8192;        // keep act chunk (<=75 MiB) L2-resident
  if (rows > M) rows = M;

  bsrbf_convert_weights<<<(512 * KTOT + 255) / 256, 256, 0, stream>>>(bw, sw, wc);

  for (long long mb = 0; mb < M; mb += rows) {
    long long cur = (M - mb < rows) ? (M - mb) : rows;
    bsrbf_act<<<(int)cur, 256, 0, stream>>>(x + mb * IN_DIM, g, b, act);
    dim3 gg((unsigned)(cur / 128), 4);
    bsrbf_gemm<<<gg, 256, 0, stream>>>(act, wc, out + mb * 512);
  }
}
